// SDM_37993280701171
// MI455X (gfx1250) — matrix operand order repaired and checked by reading, not compiled, not run on hardware
//
#include <hip/hip_runtime.h>
#include <hip/hip_bf16.h>

// ---------------------------------------------------------------------------
// Non-local attention block for MI455X (gfx1250, wave32, WMMA bf16 K=32).
// Flash-style: S (9216x9216) never materialized. Async LDS staging
// (global_load_async_to_lds_b128 / s_wait_asynccnt) for block-level operand
// reuse; per-lane online softmax stats (no per-tile shuffles).
// ---------------------------------------------------------------------------

#define CDIM  256          // channels
#define NPIX  9216         // 96*96
#define KCONV 512          // conv input channels
#define ATT_SCALE 0.0625f  // 1/sqrt(256)

typedef __attribute__((ext_vector_type(16))) __bf16 v16bf;
typedef __attribute__((ext_vector_type(8)))  __bf16 v8bf;
typedef __attribute__((ext_vector_type(8)))  float  v8f;

static __device__ __forceinline__ v8f wmma_bf16(v16bf a, v16bf b, v8f c) {
  return __builtin_amdgcn_wmma_f32_16x16x32_bf16(false, a, false, b, (short)0, c,
                                                 false, false);
}

// A-matrix 16x32 bf16: lane m = l%16, h = l>>4; elems e -> K = 16h+e
// (one contiguous 32B run).
static __device__ __forceinline__ v16bf load_A16x32(const __bf16* p, int ld) {
  int lane = threadIdx.x & 31;
  int m = lane & 15, h = lane >> 4;
  const __bf16* q = p + (size_t)m * ld + 16 * h;
  v16bf r;
  ((v8bf*)&r)[0] = *(const v8bf*)(q);
  ((v8bf*)&r)[1] = *(const v8bf*)(q + 8);
  return r;
}

// B-matrix 32x16 bf16: lane col j = l%16, h = l>>4; elems e -> K = 16h+e
// (one contiguous 32B run). Source stored K-contiguous.
static __device__ __forceinline__ v16bf load_B32x16T(const __bf16* p, int ld) {
  int lane = threadIdx.x & 31;
  int j = lane & 15, h = lane >> 4;
  const __bf16* q = p + (size_t)j * ld + 16 * h;
  v16bf r;
  ((v8bf*)&r)[0] = *(const v8bf*)(q);
  ((v8bf*)&r)[1] = *(const v8bf*)(q + 8);
  return r;
}

// CDNA5 async global->LDS copy (16B per lane), tracked by ASYNCcnt.
static __device__ __forceinline__ void async_b128(const __bf16* gptr, __bf16* lptr) {
  uint32_t ldsoff = (uint32_t)(uintptr_t)lptr;  // flat addr low 32b == LDS offset
  asm volatile("global_load_async_to_lds_b128 %0, %1, off"
               :: "v"(ldsoff), "v"((unsigned long long)(uintptr_t)gptr)
               : "memory");
}
static __device__ __forceinline__ void wait_async0() {
  asm volatile("s_wait_asynccnt 0x0" ::: "memory");
}

// ---------------------------------------------------------------------------
// Kernel 0a: straight fp32 -> bf16 convert
__global__ void k_cvt(const float* __restrict__ src, __bf16* __restrict__ dst, int n) {
  int i = blockIdx.x * blockDim.x + threadIdx.x;
  if (i < n) dst[i] = (__bf16)src[i];
}

// Kernel 0b: fp32 [rows, cols] -> bf16 [cols, rows] (transpose + convert)
__global__ void k_cvt_t(const float* __restrict__ src, __bf16* __restrict__ dst,
                        int rows, int cols) {
  int i = blockIdx.x * blockDim.x + threadIdx.x;
  if (i < rows * cols) {
    int r = i / cols, c = i % cols;
    dst[(size_t)c * rows + r] = (__bf16)src[i];
  }
}

// ---------------------------------------------------------------------------
// Kernel 1: softmax row stats. Block = 8 waves = 128 rows of S. The ftT
// m-chunk (32 cols x 256 c = 16KB) is async-staged into LDS and shared by all
// 8 waves; per-wave A panel in 64 VGPRs. Stats are per-lane online (each lane
// covers a fixed column residue class) -> no shuffles in the hot loop; one
// width-16 merge at the end. Stores rowmax and 1/rowsum.
#define ST_MCHUNK 32
__global__ void __launch_bounds__(256) k_stats(const __bf16* __restrict__ fmT,
                                               const __bf16* __restrict__ ftT,
                                               float* __restrict__ rmax,
                                               float* __restrict__ rsum_inv) {
  __shared__ __bf16 stage[2][ST_MCHUNK * CDIM];  // 2 x 16KB
  const int w = threadIdx.x >> 5;
  const int lane = threadIdx.x & 31;
  const int h = lane >> 4;
  const int n0 = blockIdx.x * 128 + w * 16;
  const int tid = threadIdx.x;

  v16bf A[8];
#pragma unroll
  for (int kc = 0; kc < 8; ++kc)
    A[kc] = load_A16x32(fmT + (size_t)n0 * CDIM + kc * 32, CDIM);

  float rm[8], rs[8];
#pragma unroll
  for (int r = 0; r < 8; ++r) { rm[r] = -1e30f; rs[r] = 0.0f; }

  const int NCH = NPIX / ST_MCHUNK;  // 288
  {  // prime chunk 0: 256 threads x 64B (4 async B128s each)
    const __bf16* g = ftT + (size_t)tid * 32;
    __bf16* l = &stage[0][(size_t)tid * 32];
#pragma unroll
    for (int k = 0; k < 4; ++k) async_b128(g + k * 8, l + k * 8);
  }
  for (int nc = 0; nc < NCH; ++nc) {
    const int cur = nc & 1, nxt = cur ^ 1;
    wait_async0();
    __syncthreads();  // stage[cur] ready for everyone
    if (nc + 1 < NCH) {
      const __bf16* g = ftT + (size_t)(nc + 1) * ST_MCHUNK * CDIM + (size_t)tid * 32;
      __bf16* l = &stage[nxt][(size_t)tid * 32];
#pragma unroll
      for (int k = 0; k < 4; ++k) async_b128(g + k * 8, l + k * 8);
    }
#pragma unroll
    for (int mt = 0; mt < 2; ++mt) {
      v8f acc = {};
#pragma unroll
      for (int kc = 0; kc < 8; ++kc) {
        v16bf B = load_B32x16T(&stage[cur][(size_t)(mt * 16) * CDIM + kc * 32], CDIM);
        acc = wmma_bf16(A[kc], B, acc);
      }
      // Per-lane online update: 2 exp + max + fma per element, no shuffles.
#pragma unroll
      for (int r = 0; r < 8; ++r) {
        float v = acc[r] * ATT_SCALE;
        float nm = fmaxf(rm[r], v);
        rs[r] = rs[r] * __expf(rm[r] - nm) + __expf(v - nm);
        rm[r] = nm;
      }
    }
  }
  // Merge the 16 per-lane (max, sum) pairs within each half (width-16 xor).
#pragma unroll
  for (int d = 1; d < 16; d <<= 1) {
#pragma unroll
    for (int r = 0; r < 8; ++r) {
      float rm2 = __shfl_xor(rm[r], d, 16);
      float rs2 = __shfl_xor(rs[r], d, 16);
      float nm = fmaxf(rm[r], rm2);
      rs[r] = rs[r] * __expf(rm[r] - nm) + rs2 * __expf(rm2 - nm);
      rm[r] = nm;
    }
  }
  if ((lane & 15) == 0) {
#pragma unroll
    for (int r = 0; r < 8; ++r) {
      rmax[n0 + r + 8 * h] = rm[r];
      rsum_inv[n0 + r + 8 * h] = 1.0f / rs[r];
    }
  }
}

// ---------------------------------------------------------------------------
// Kernel 2: cooperative flash PV. Block = 8 waves owns a 64-wide m strip and
// all 256 channels. Per n-chunk (32): fmT chunk async-staged to LDS; wave w
// computes S quadrant (t = w>>2, mq = w&3) with its register-hoisted B panel,
// normalizes -> bf16 P into shared LDS; then every wave accumulates its
// 2 c-tiles x 4 m-tiles (64 acc VGPRs) from fmB x P (A panel hoisted per chunk).
#define PV_MW 64
__global__ void __launch_bounds__(256) k_pv(const __bf16* __restrict__ fmT,
                                            const __bf16* __restrict__ ftT,
                                            const __bf16* __restrict__ fmB,
                                            const float* __restrict__ rmax,
                                            const float* __restrict__ rsum_inv,
                                            __bf16* __restrict__ attnT) {
  __shared__ __bf16 stage[2][32 * CDIM];     // fmT n-chunk, 2 x 16KB
  __shared__ __bf16 Pbuf[PV_MW * 32];        // P block [m][n], 4KB
  const int w = threadIdx.x >> 5;
  const int lane = threadIdx.x & 31;
  const int j = lane & 15, h = lane >> 4;
  const int tid = threadIdx.x;
  const int m0b = blockIdx.x * PV_MW;
  const int t = w >> 2;    // n-subtile for S phase
  const int mq = w & 3;    // m-tile for S phase

  v16bf Bs[8];
#pragma unroll
  for (int kc = 0; kc < 8; ++kc)
    Bs[kc] = load_B32x16T(ftT + (size_t)(m0b + mq * 16) * CDIM + kc * 32, CDIM);

  v8f acc[2][4];  // [c-tile within wave][m-tile]
#pragma unroll
  for (int ci = 0; ci < 2; ++ci)
#pragma unroll
    for (int mt = 0; mt < 4; ++mt) acc[ci][mt] = (v8f){};

  const int NCH = NPIX / 32;  // 288
  {  // prime chunk 0 (16KB, 64B per thread)
    const __bf16* g = fmT + (size_t)tid * 32;
    __bf16* l = &stage[0][(size_t)tid * 32];
#pragma unroll
    for (int k = 0; k < 4; ++k) async_b128(g + k * 8, l + k * 8);
  }
  for (int nc = 0; nc < NCH; ++nc) {
    const int cur = nc & 1, nxt = cur ^ 1;
    const int n0 = nc * 32;
    wait_async0();
    __syncthreads();  // stage[cur] ready; previous PV reads of Pbuf done

    // ---- S phase: this wave's quadrant (rows n0+16t.., cols m0b+16mq..)
    v8f s = {};
#pragma unroll
    for (int kc = 0; kc < 8; ++kc) {
      v16bf As = load_A16x32(&stage[cur][(size_t)(16 * t) * CDIM + kc * 32], CDIM);
      s = wmma_bf16(As, Bs[kc], s);
    }
#pragma unroll
    for (int r = 0; r < 8; ++r) {
      int n = n0 + 16 * t + r + 8 * h;
      float p = __expf(s[r] * ATT_SCALE - rmax[n]) * rsum_inv[n];
      Pbuf[(size_t)(mq * 16 + j) * 32 + (16 * t + r + 8 * h)] = (__bf16)p;
    }
    __syncthreads();  // publish P

    // ---- PV phase: prefetch next fmT chunk, hoist A panel, accumulate.
    if (nc + 1 < NCH) {
      const __bf16* g = fmT + (size_t)(n0 + 32) * CDIM + (size_t)tid * 32;
      __bf16* l = &stage[nxt][(size_t)tid * 32];
#pragma unroll
      for (int k = 0; k < 4; ++k) async_b128(g + k * 8, l + k * 8);
    }
    v16bf A2[2];  // invariant across the 4 m-tiles -> load once per chunk
#pragma unroll
    for (int ci = 0; ci < 2; ++ci)
      A2[ci] = load_A16x32(fmB + (size_t)((w * 2 + ci) * 16) * NPIX + n0, NPIX);
#pragma unroll
    for (int mt = 0; mt < 4; ++mt) {
      v16bf Pb;  // B-layout read: lane j, K = 16h + e (contiguous 32B)
      const __bf16* q = &Pbuf[(size_t)(mt * 16 + j) * 32 + 16 * h];
      ((v8bf*)&Pb)[0] = *(const v8bf*)(q);
      ((v8bf*)&Pb)[1] = *(const v8bf*)(q + 8);
#pragma unroll
      for (int ci = 0; ci < 2; ++ci)
        acc[ci][mt] = wmma_bf16(A2[ci], Pb, acc[ci][mt]);
    }
  }
  // Store transposed bf16 attnT[M, C]: lane col m fixed, 8 consecutive c.
#pragma unroll
  for (int ci = 0; ci < 2; ++ci) {
    int ct = w * 2 + ci;
#pragma unroll
    for (int mt = 0; mt < 4; ++mt) {
      v8bf ob;
#pragma unroll
      for (int r = 0; r < 8; ++r) ob[r] = (__bf16)acc[ci][mt][r];
      *(v8bf*)(attnT + (size_t)(m0b + mt * 16 + j) * CDIM + ct * 16 + 8 * h) = ob;
    }
  }
}

// ---------------------------------------------------------------------------
// Kernel 3: 1x1 conv = GEMM [256 x 9216 x 512]. Each wave holds its full
// theta panel (16 o x 512 k = 128 VGPRs) and loops 36 p-tiles; the block's
// 4 waves (4 consecutive o-tiles) share an async-staged B panel in LDS
// (attnT rows (k<256) ⊕ ftT rows (k>=256), 16KB per p-tile, double-buffered).
#define CV_PT 36  // p-tiles per block (576 / 16 p-groups)
__global__ void __launch_bounds__(128) k_conv(const __bf16* __restrict__ thetaB,
                                              const __bf16* __restrict__ attnT,
                                              const __bf16* __restrict__ ftT,
                                              const float* __restrict__ bias,
                                              float* __restrict__ out) {
  __shared__ __bf16 bstage[2][16 * KCONV];  // 2 x 16KB: [p][k]
  const int w = threadIdx.x >> 5;
  const int lane = threadIdx.x & 31;
  const int j = lane & 15, h = lane >> 4;
  const int tid = threadIdx.x;
  const int og = blockIdx.x & 3;   // o-group (4 o-tiles per block)
  const int pg = blockIdx.x >> 2;  // p-group (36 p-tiles each)
  const int o0 = (og * 4 + w) * 16;

  v16bf A[16];  // full theta panel for this wave's 16 output channels
#pragma unroll
  for (int kc = 0; kc < 16; ++kc)
    A[kc] = load_A16x32(thetaB + (size_t)o0 * KCONV + kc * 32, KCONV);
  float bv[8];
#pragma unroll
  for (int r = 0; r < 8; ++r) bv[r] = bias[o0 + r + 8 * h];

  // Stage helper: thread tid copies 128B of the [16 p][512 k] panel; each
  // 128B block lies entirely in the attnT half (k<256) or ftT half.
  auto issue_stage = [&](int p0, __bf16* dstBase) {
    int off = tid * 64;               // element offset (128B = 64 bf16)
    int p = off >> 9;                 // 512 elems per p row
    int rem = off & 511;
    const __bf16* g = (rem < 256) ? (attnT + (size_t)(p0 + p) * CDIM + rem)
                                  : (ftT + (size_t)(p0 + p) * CDIM + (rem - 256));
    __bf16* l = dstBase + off;
#pragma unroll
    for (int k = 0; k < 8; ++k) async_b128(g + k * 8, l + k * 8);
  };

  const int pt0 = pg * CV_PT;
  issue_stage(pt0 * 16, &bstage[0][0]);
  for (int it = 0; it < CV_PT; ++it) {
    const int cur = it & 1, nxt = cur ^ 1;
    const int p0 = (pt0 + it) * 16;
    wait_async0();
    __syncthreads();
    if (it + 1 < CV_PT) issue_stage(p0 + 16, &bstage[nxt][0]);
    v8f acc = {};
#pragma unroll
    for (int kc = 0; kc < 16; ++kc) {
      v16bf B = load_B32x16T(&bstage[cur][kc * 32], KCONV);
      acc = wmma_bf16(A[kc], B, acc);
    }
#pragma unroll
    for (int r = 0; r < 8; ++r)
      out[(size_t)(o0 + r + 8 * h) * NPIX + p0 + j] = acc[r] + bv[r];
  }
}

// ---------------------------------------------------------------------------
extern "C" void kernel_launch(void* const* d_in, const int* in_sizes, int n_in,
                              void* d_out, int out_size, void* d_ws, size_t ws_size,
                              hipStream_t stream) {
  const float* fm = (const float*)d_in[0];        // [256, 9216]
  const float* ft = (const float*)d_in[1];        // [256, 9216]
  const float* tw = (const float*)d_in[2];        // [256, 512]
  const float* tb = (const float*)d_in[3];        // [256]
  float* out = (float*)d_out;                     // [256, 9216]

  char* ws = (char*)d_ws;
  size_t off = 0;
  auto walloc = [&](size_t bytes) -> void* {
    void* p = ws + off;
    off = (off + bytes + 255) & ~(size_t)255;
    return p;
  };
  __bf16* fmT    = (__bf16*)walloc((size_t)NPIX * CDIM * 2);  // [N, C]
  __bf16* ftT    = (__bf16*)walloc((size_t)NPIX * CDIM * 2);  // [M, C]
  __bf16* fmB    = (__bf16*)walloc((size_t)CDIM * NPIX * 2);  // [C, N]
  __bf16* thetaB = (__bf16*)walloc((size_t)CDIM * KCONV * 2); // [O, 512]
  __bf16* attnT  = (__bf16*)walloc((size_t)NPIX * CDIM * 2);  // [M, C]
  float*  rmax   = (float*)walloc((size_t)NPIX * 4);
  float*  rsumI  = (float*)walloc((size_t)NPIX * 4);
  (void)ws_size; (void)in_sizes; (void)n_in; (void)out_size;

  const int nElem = CDIM * NPIX;  // 2,359,296
  k_cvt_t<<<(nElem + 255) / 256, 256, 0, stream>>>(fm, fmT, CDIM, NPIX);
  k_cvt_t<<<(nElem + 255) / 256, 256, 0, stream>>>(ft, ftT, CDIM, NPIX);
  k_cvt  <<<(nElem + 255) / 256, 256, 0, stream>>>(fm, fmB, nElem);
  k_cvt  <<<(CDIM * KCONV + 255) / 256, 256, 0, stream>>>(tw, thetaB, CDIM * KCONV);

  // 576 row-tiles / 8 waves per block
  k_stats<<<72, 256, 0, stream>>>(fmT, ftT, rmax, rsumI);
  // 144 blocks x 64-wide m strips (8 cooperative waves)
  k_pv<<<144, 256, 0, stream>>>(fmT, ftT, fmB, rmax, rsumI, attnT);
  // 64 blocks: 4 o-groups x 16 p-groups, 36 p-tiles each
  k_conv<<<64, 128, 0, stream>>>(thetaB, attnT, ftT, tb, out);
}
